// Gps_26749056319728
// MI455X (gfx1250) — matrix operand order repaired and checked by reading, not compiled, not run on hardware
//
#include <hip/hip_runtime.h>
#include <hip/hip_bf16.h>

// ---------------------------------------------------------------------------
// CDNA5 (gfx1250) pipeline for the 2-layer GraphSAGE-style encoder.
//   Bp1/Bp2/Wp1/Wp2 : one-time f32 -> bf16 repack into WMMA-fragment layout
//   agg0 = 0.5 * adj1 @ fl1   (TDM-staged A tiles in LDS + WMMA bf16)
//   agg1 = 0.5 * adj2 @ fl2   (TDM-staged A tiles in LDS + WMMA bf16)
//   h    = lrelu([gather(fm,seed), agg0] @ W1^T)   (fused, packed W)
//   emb  = lrelu([h, mean4(agg1)]   @ W2^T)        (fused, packed W)
// ---------------------------------------------------------------------------

typedef __attribute__((ext_vector_type(16))) __bf16 v16bf;
typedef __attribute__((ext_vector_type(8)))  __bf16 v8bf;
typedef __attribute__((ext_vector_type(8)))  float  v8f;
typedef __attribute__((ext_vector_type(4)))  unsigned int uint32x4;
typedef __attribute__((ext_vector_type(8)))  int int32x8;
typedef __attribute__((ext_vector_type(4)))  int int32x4;

__device__ __forceinline__ v8f wmma_bf16(v16bf a, v16bf b, v8f c) {
  return __builtin_amdgcn_wmma_f32_16x16x32_bf16(
      /*neg_a=*/false, a, /*neg_b=*/false, b,
      /*c_mod=*/(short)0, c, /*reuse_a=*/false, /*reuse_b=*/false);
}

// A fragment (16x32 bf16): lane = row m (lane&15); 16 consecutive k
// at kb=(lane>=16)*16.  ap -> source row + kb (16B aligned;
// works for global or LDS pointers).
__device__ __forceinline__ v16bf pack_a(const float* ap) {
  const float4 x0 = *reinterpret_cast<const float4*>(ap);
  const float4 x1 = *reinterpret_cast<const float4*>(ap + 4);
  const float4 y0 = *reinterpret_cast<const float4*>(ap + 8);
  const float4 y1 = *reinterpret_cast<const float4*>(ap + 12);
  v16bf a;
  a[0]  = (__bf16)x0.x; a[1]  = (__bf16)x0.y; a[2]  = (__bf16)x0.z; a[3]  = (__bf16)x0.w;
  a[4]  = (__bf16)x1.x; a[5]  = (__bf16)x1.y; a[6]  = (__bf16)x1.z; a[7]  = (__bf16)x1.w;
  a[8]  = (__bf16)y0.x; a[9]  = (__bf16)y0.y; a[10] = (__bf16)y0.z; a[11] = (__bf16)y0.w;
  a[12] = (__bf16)y1.x; a[13] = (__bf16)y1.y; a[14] = (__bf16)y1.z; a[15] = (__bf16)y1.w;
  return a;
}

// A fragment averaging 4 consecutive rows (stride 256): fuses the topn=4
// segment-mean of agg1 into the layer-2 operand build.
__device__ __forceinline__ v16bf pack_a_mean4(const float* p) {
  v16bf a;
#pragma unroll
  for (int j = 0; j < 8; ++j) {
    float s0 = 0.25f * (p[j]      + p[256 + j]      + p[512 + j]      + p[768 + j]);
    float s1 = 0.25f * (p[8 + j] + p[256 + 8 + j] + p[512 + 8 + j] + p[768 + 8 + j]);
    a[j]     = (__bf16)s0;
    a[8 + j] = (__bf16)s1;
  }
  return a;
}

// ---------------------------------------------------------------------------
// Tensor Data Mover: DMA a 32-row x 128-col f32 tile of A (row stride K) into
// LDS at lds_byte_off, with 4-dword padding after every 32 dwords (LDS row
// stride becomes 144 dwords) to spread WMMA-fragment reads across banks.
// D# layout per CDNA5 ISA.
// ---------------------------------------------------------------------------
#define TDM_TILE_K 128
#define LDS_ROW    144   // 128 data dwords + 4x4 pad dwords

__device__ __forceinline__ void tdm_issue_2d(const float* gsrc, unsigned lds_byte_off,
                                             int K, int M) {
  const unsigned long long ga = (unsigned long long)(uintptr_t)gsrc;
  uint32x4 g0;
  g0[0] = 1u;                                    // count=1 (valid), user mode
  g0[1] = lds_byte_off;                          // lds_addr (bytes)
  g0[2] = (unsigned)ga;                          // global_addr[31:0]
  g0[3] = (unsigned)(ga >> 32) | 0x80000000u;    // global_addr[56:32] | type=2
  int32x8 g1;
  g1[0] = (2 << 16)       // data_size = 4B
        | (1 << 20)       // pad_enable
        | (4 << 22)       // pad_interval: every 32 dwords
        | (3 << 25);      // pad_amount: 4 dwords
  g1[1] = (int)(((unsigned)K & 0xffffu) << 16);                       // tensor_dim0 lo
  g1[2] = (int)(((unsigned)K >> 16) | (((unsigned)M & 0xffffu) << 16)); // dim0 hi | dim1 lo
  g1[3] = (int)(((unsigned)M >> 16) | (128u << 16));                  // dim1 hi | tile_dim0=128
  g1[4] = 32;                                                          // tile_dim1=32, tile_dim2=0
  g1[5] = K;                                                           // tensor_dim0_stride lo
  g1[6] = 0;
  g1[7] = 0;
  const int32x4 z4 = {0, 0, 0, 0};
#if __clang_major__ >= 23
  const int32x8 z8 = {0, 0, 0, 0, 0, 0, 0, 0};
  __builtin_amdgcn_tensor_load_to_lds(g0, g1, z4, z4, z8, 0);
#else
  __builtin_amdgcn_tensor_load_to_lds(g0, g1, z4, z4, 0);
#endif
}

// ---------------------------------------------------------------------------
// One-time operand repacks (all tiny and L2-resident afterwards).
// Bp layout: Bp[k/32][n][kk] = B[k][n]  -> a WMMA B fragment is 16 contiguous
// bf16 (32B) per lane.  Wp: straight f32->bf16 of row-major W[256,512].
// ---------------------------------------------------------------------------
__global__ __launch_bounds__(256)
void pack_b_kernel(const float* __restrict__ B, __bf16* __restrict__ Bp) {
  const int t = blockIdx.x;       // 32-k slab
  const int n = threadIdx.x;      // column 0..255
  const float* src = B + (size_t)t * 32 * 256 + n;
  v16bf lo, hi;
#pragma unroll
  for (int j = 0; j < 16; ++j) {
    lo[j] = (__bf16)src[(size_t)j * 256];
    hi[j] = (__bf16)src[(size_t)(j + 16) * 256];
  }
  v16bf* dst = (v16bf*)(Bp + ((size_t)t * 256 + n) * 32);
  dst[0] = lo;
  dst[1] = hi;
}

__global__ __launch_bounds__(256)
void cvt_bf16_kernel(const float* __restrict__ X, __bf16* __restrict__ Y) {
  const size_t i = ((size_t)blockIdx.x * 256 + threadIdx.x) * 8;
  const float4 a = *reinterpret_cast<const float4*>(X + i);
  const float4 b = *reinterpret_cast<const float4*>(X + i + 4);
  v8bf y;
  y[0] = (__bf16)a.x; y[1] = (__bf16)a.y; y[2] = (__bf16)a.z; y[3] = (__bf16)a.w;
  y[4] = (__bf16)b.x; y[5] = (__bf16)b.y; y[6] = (__bf16)b.z; y[7] = (__bf16)b.w;
  *reinterpret_cast<v8bf*>(Y + i) = y;
}

// ---------------------------------------------------------------------------
// C[M,256] = scale * (A[M,K] @ B[K,256]).  Block = 32 rows x all 256 cols,
// 8 waves; wave (wm,wn) owns rows wm*16.. and cols wn*64.. via 4 accumulators.
// A tiles stream HBM -> LDS via TDM (double buffered); B fragments are 32B
// v16bf loads from the packed bf16 Bp (L2-resident).
// ---------------------------------------------------------------------------
__global__ __launch_bounds__(256)
void gemm_scale_tdm(const float* __restrict__ A, const __bf16* __restrict__ Bp,
                    float* __restrict__ C, int M, int K, float scale) {
  __shared__ float atile[2][32 * LDS_ROW];

  const int lane = threadIdx.x & 31;
  const int wave = threadIdx.x >> 5;
  const int wm = wave >> 2;        // 0..1
  const int wn = wave & 3;         // 0..3
  const int m    = lane & 15;
  const int half = lane >> 4;
  const int kb8  = half << 4;
  const int row0 = blockIdx.x * 32;

  if (wave == 0) {
    tdm_issue_2d(A + (size_t)row0 * K,
                 (unsigned)(uintptr_t)&atile[0][0], K, M);
  }

  v8f acc0 = {}, acc1 = {}, acc2 = {}, acc3 = {};
  const int ntiles = K / TDM_TILE_K;
  for (int t = 0; t < ntiles; ++t) {
    __syncthreads();   // readers of atile[(t+1)&1] (iteration t-1) are done
    if (wave == 0) {
      if (t + 1 < ntiles) {
        tdm_issue_2d(A + (size_t)row0 * K + (size_t)(t + 1) * TDM_TILE_K,
                     (unsigned)(uintptr_t)&atile[(t + 1) & 1][0], K, M);
        __builtin_amdgcn_s_wait_tensorcnt(1);   // in-order: tile t has landed
      } else {
        __builtin_amdgcn_s_wait_tensorcnt(0);
      }
    }
    __syncthreads();   // tile t visible to all waves
    const float* arow = &atile[t & 1][(wm * 16 + m) * LDS_ROW] + kb8;
#pragma unroll
    for (int c = 0; c < 4; ++c) {
      const v16bf a = pack_a(arow + c * 36);    // ds_load_b128 x4 + pk-cvt
      const v16bf* bv =
          (const v16bf*)(Bp + (size_t)(t * 4 + c) * (256 * 32));
      const v16bf b0 = bv[(wn * 64 +  0 + m) * 2 + half];
      const v16bf b1 = bv[(wn * 64 + 16 + m) * 2 + half];
      const v16bf b2 = bv[(wn * 64 + 32 + m) * 2 + half];
      const v16bf b3 = bv[(wn * 64 + 48 + m) * 2 + half];
      acc0 = wmma_bf16(a, b0, acc0);
      acc1 = wmma_bf16(a, b1, acc1);
      acc2 = wmma_bf16(a, b2, acc2);
      acc3 = wmma_bf16(a, b3, acc3);
    }
  }

  float* cp = C + (size_t)(row0 + wm * 16 + half * 8) * 256 + wn * 64 + m;
#pragma unroll
  for (int r = 0; r < 8; ++r) {
    cp[(size_t)r * 256 +  0] = acc0[r] * scale;
    cp[(size_t)r * 256 + 16] = acc1[r] * scale;
    cp[(size_t)r * 256 + 32] = acc2[r] * scale;
    cp[(size_t)r * 256 + 48] = acc3[r] * scale;
  }
}

// ---------------------------------------------------------------------------
// out[M,256] = lrelu(concat(X1', X2') @ W^T), Wp = bf16 row-major [256,512].
//   seed != nullptr : X1' row i = X1[seed[i]]         (layer-1 gather)
//   mean4 != 0      : X2' row i = mean(X2[4i..4i+3])  (layer-2 segment mean)
// ---------------------------------------------------------------------------
__global__ __launch_bounds__(256)
void fused_encoder_kernel(const float* __restrict__ X1, const int* __restrict__ seed,
                          const float* __restrict__ X2, int mean4,
                          const __bf16* __restrict__ Wp, float* __restrict__ out) {
  const int lane = threadIdx.x & 31;
  const int wave = threadIdx.x >> 5;
  const int wm = wave >> 2;
  const int wn = wave & 3;
  const int row0 = blockIdx.x * 32 + wm * 16;
  const int m    = lane & 15;
  const int half = lane >> 4;
  const int r    = row0 + m;

  const float* x1p = (seed ? (X1 + (size_t)seed[r] * 256)
                           : (X1 + (size_t)r * 256)) + (half << 4);
  const float* x2p = (mean4 ? (X2 + ((size_t)r << 2) * 256)
                            : (X2 + (size_t)r * 256)) + (half << 4);

  const v16bf* wv = (const v16bf*)Wp;      // 32 v16bf per 512-wide row
  const int c0 = wn * 64 + m;

  v8f acc0 = {}, acc1 = {}, acc2 = {}, acc3 = {};
#pragma unroll
  for (int k0 = 0; k0 < 512; k0 += 32) {
    v16bf a;
    if (k0 < 256) {
      a = pack_a(x1p + k0);
    } else if (!mean4) {
      a = pack_a(x2p + (k0 - 256));
    } else {
      a = pack_a_mean4(x2p + (k0 - 256));
    }
    const int kv = (k0 >> 4) + half;
    const v16bf b0 = wv[(size_t)(c0 +  0) * 32 + kv];
    const v16bf b1 = wv[(size_t)(c0 + 16) * 32 + kv];
    const v16bf b2 = wv[(size_t)(c0 + 32) * 32 + kv];
    const v16bf b3 = wv[(size_t)(c0 + 48) * 32 + kv];
    acc0 = wmma_bf16(a, b0, acc0);
    acc1 = wmma_bf16(a, b1, acc1);
    acc2 = wmma_bf16(a, b2, acc2);
    acc3 = wmma_bf16(a, b3, acc3);
  }

  float* cp = out + (size_t)(row0 + half * 8) * 256 + wn * 64 + m;
#pragma unroll
  for (int rr = 0; rr < 8; ++rr) {
    float v0 = acc0[rr], v1 = acc1[rr], v2 = acc2[rr], v3 = acc3[rr];
    cp[(size_t)rr * 256 +  0] = v0 >= 0.f ? v0 : 0.01f * v0;
    cp[(size_t)rr * 256 + 16] = v1 >= 0.f ? v1 : 0.01f * v1;
    cp[(size_t)rr * 256 + 32] = v2 >= 0.f ? v2 : 0.01f * v2;
    cp[(size_t)rr * 256 + 48] = v3 >= 0.f ? v3 : 0.01f * v3;
  }
}

// ---------------------------------------------------------------------------
extern "C" void kernel_launch(void* const* d_in, const int* in_sizes, int n_in,
                              void* d_out, int out_size, void* d_ws, size_t ws_size,
                              hipStream_t stream) {
  (void)in_sizes; (void)n_in; (void)out_size; (void)ws_size;

  const float* fm   = (const float*)d_in[0];  // [100000, 256]
  const float* adj1 = (const float*)d_in[1];  // [16384, 4096]
  const float* fl1  = (const float*)d_in[2];  // [4096, 256]
  const float* adj2 = (const float*)d_in[3];  // [65536, 2048]
  const float* fl2  = (const float*)d_in[4];  // [2048, 256]
  const float* W1   = (const float*)d_in[5];  // [256, 512]
  const float* W2   = (const float*)d_in[6];  // [256, 512]
  const int*   seed = (const int*)d_in[7];    // [16384]
  float* out = (float*)d_out;                 // [16384, 256]

  constexpr int N_SEED = 16384;
  constexpr int M2     = 65536;               // N_SEED * topn

  char* ws = (char*)d_ws;
  float*  agg0 = (float*)(ws);                          // 16 MB
  float*  agg1 = (float*)(ws + (size_t)16  * (1 << 20)); // 64 MB
  float*  h    = (float*)(ws + (size_t)80  * (1 << 20)); // 16 MB
  __bf16* Bp1  = (__bf16*)(ws + (size_t)96 * (1 << 20)); //  2 MB
  __bf16* Bp2  = (__bf16*)(ws + (size_t)98 * (1 << 20)); //  1 MB
  __bf16* Wp1  = (__bf16*)(ws + (size_t)99 * (1 << 20)); // 256 KB
  __bf16* Wp2  = (__bf16*)(ws + (size_t)99 * (1 << 20) + 256 * 512 * 2);

  // one-time bf16 repacks (tiny)
  pack_b_kernel<<<4096 / 32, 256, 0, stream>>>(fl1, Bp1);
  pack_b_kernel<<<2048 / 32, 256, 0, stream>>>(fl2, Bp2);
  cvt_bf16_kernel<<<(256 * 512 / 8) / 256, 256, 0, stream>>>(W1, Wp1);
  cvt_bf16_kernel<<<(256 * 512 / 8) / 256, 256, 0, stream>>>(W2, Wp2);

  // step 1a: agg0 = 0.5 * adj1 @ fl1
  gemm_scale_tdm<<<N_SEED / 32, 256, 0, stream>>>(adj1, Bp1, agg0, N_SEED, 4096, 0.5f);
  // step 2a: agg1 = 0.5 * adj2 @ fl2
  gemm_scale_tdm<<<M2 / 32, 256, 0, stream>>>(adj2, Bp2, agg1, M2, 2048, 0.5f);
  // step 1b: h = lrelu([fm[seed], agg0] @ W1^T)
  fused_encoder_kernel<<<N_SEED / 32, 256, 0, stream>>>(fm, seed, agg0, 0, Wp1, h);
  // step 2b: emb = lrelu([h, mean4(agg1)] @ W2^T)
  fused_encoder_kernel<<<N_SEED / 32, 256, 0, stream>>>(h, nullptr, agg1, 1, Wp2, out);
}
